// GNN_model_14499809592153
// MI455X (gfx1250) — compile-verified, matrix operand order checked by reading, not run on hardware
//
#include <hip/hip_runtime.h>

// ---------------------------------------------------------------------------
// GraphSAGE layer (mean aggregate -> agg@W_l + x@W_r + b -> ReLU) for MI455X.
//
// Roofline: GEMM phase moves ~77 MB (3.3 us @ 23.3 TB/s). BF16 WMMA
// (v_wmma_f32_16x16x32_bf16, fp32 accumulate) keeps matrix math far below the
// bandwidth floor; fp32 WMMA (16x16x4) would not. Scatter phase is L2-atomic
// bound (x = 25.6 MB lives in the 192 MB L2), so we use hardware
// global_atomic_add_f32 via __hip_atomic_fetch_add.
// ---------------------------------------------------------------------------

#define N_NODES 50000
#define N_EDGES 600000
#define D       128
#define M_TILES (N_NODES / 16)     // 3125, exact
#define WAVES_PER_BLOCK 8
#define ROWS_PER_BLOCK  128
#define LDS_STRIDE      136        // 128 + 8 halfword pad (row base stays 16B aligned: 272 = 16*17)

typedef __attribute__((ext_vector_type(16))) __bf16 v16bf;
typedef __attribute__((ext_vector_type(8)))  __bf16 v8bf;
typedef __attribute__((ext_vector_type(8)))  float  v8f;

__device__ __forceinline__ unsigned bf16_1(float x) {
  // round-to-nearest-even fp32 -> bf16
  unsigned u = __float_as_uint(x);
  return (u + 0x7FFFu + ((u >> 16) & 1u)) >> 16;
}
__device__ __forceinline__ unsigned bf16_2(float lo, float hi) {
  return bf16_1(lo) | (bf16_1(hi) << 16);
}

// --------------------------- kernel 1: zero workspace ----------------------
__global__ void k_zero_f4(float4* __restrict__ p, int n4) {
  int i = blockIdx.x * blockDim.x + threadIdx.x;
  if (i < n4) p[i] = make_float4(0.f, 0.f, 0.f, 0.f);
}

// --------------------------- kernel 2: edge scatter ------------------------
// One wave (32 lanes) per edge: coalesced 512B gather of x[src], fp32 atomics
// into msg[dst]; lane 0 bumps degree.
__global__ __launch_bounds__(256) void k_scatter(const float* __restrict__ x,
                                                 const long long* __restrict__ ei,
                                                 float* __restrict__ msg,
                                                 float* __restrict__ deg) {
  int t    = blockIdx.x * 256 + threadIdx.x;
  int e    = t >> 5;
  int lane = t & 31;
  if (e >= N_EDGES) return;
  long long src = ei[e];
  long long dst = ei[N_EDGES + e];
  const float4 v = *(const float4*)(x + src * (long long)D + lane * 4);
  float* m = msg + dst * (long long)D + lane * 4;
  __hip_atomic_fetch_add(m + 0, v.x, __ATOMIC_RELAXED, __HIP_MEMORY_SCOPE_AGENT);
  __hip_atomic_fetch_add(m + 1, v.y, __ATOMIC_RELAXED, __HIP_MEMORY_SCOPE_AGENT);
  __hip_atomic_fetch_add(m + 2, v.z, __ATOMIC_RELAXED, __HIP_MEMORY_SCOPE_AGENT);
  __hip_atomic_fetch_add(m + 3, v.w, __ATOMIC_RELAXED, __HIP_MEMORY_SCOPE_AGENT);
  if (lane == 0)
    __hip_atomic_fetch_add(deg + dst, 1.0f, __ATOMIC_RELAXED, __HIP_MEMORY_SCOPE_AGENT);
}

// ---------------- kernel 3: fused normalize + dual GEMM + bias + relu ------
// Block = 8 waves = 128 rows. Wave w computes rows [blk*128 + w*16, +16),
// all 128 output columns: acc[nt] += A_agg x W_l + A_x x W_r via
// v_wmma_f32_16x16x32_bf16 (8 N-tiles x 4 K-tiles x 2 matrices = 64 WMMA).
__global__ __launch_bounds__(256, 1) void k_sage_wmma(
    const float* __restrict__ x, const float* __restrict__ msg,
    const float* __restrict__ deg, const float* __restrict__ Wl,
    const float* __restrict__ Wr, const float* __restrict__ bias,
    float* __restrict__ out) {
  // 4 x 34816 B = 136 KB LDS (<= 320 KB/WGP; 2 blocks resident per WGP)
  __shared__ unsigned short sWl[D * LDS_STRIDE];               // W_l^T [n][k] bf16
  __shared__ unsigned short sWr[D * LDS_STRIDE];               // W_r^T [n][k] bf16
  __shared__ unsigned short sAg[ROWS_PER_BLOCK * LDS_STRIDE];  // agg tile bf16
  __shared__ unsigned short sAx[ROWS_PER_BLOCK * LDS_STRIDE];  // x   tile bf16

  const int t    = threadIdx.x;
  const int row0 = blockIdx.x * ROWS_PER_BLOCK;

  // --- stage weights, transposed, fp32 -> bf16 (read coalesced over n) ---
#pragma unroll 4
  for (int i = 0; i < 64; ++i) {
    int idx = i * 256 + t;             // 0..16383
    int k = idx >> 7, n = idx & 127;
    sWl[n * LDS_STRIDE + k] = (unsigned short)bf16_1(Wl[idx]);
    sWr[n * LDS_STRIDE + k] = (unsigned short)bf16_1(Wr[idx]);
  }

  // --- stage activations: agg = msg * 1/max(deg,1) (fused), plus x ---
#pragma unroll
  for (int i = 0; i < 16; ++i) {
    int idx4 = i * 256 + t;            // 4096 float4 = 128 rows x 128 cols
    int r = idx4 >> 5;                 // 0..127
    int c = (idx4 & 31) * 4;           // 0..124
    int grow = row0 + r;
    float4 vx = make_float4(0.f, 0.f, 0.f, 0.f), vm = vx;
    float inv = 0.f;
    if (grow < N_NODES) {
      vx  = *(const float4*)(x   + grow * (long long)D + c);
      vm  = *(const float4*)(msg + grow * (long long)D + c);
      inv = 1.0f / fmaxf(deg[grow], 1.0f);
    }
    uint2 px, pm;
    px.x = bf16_2(vx.x, vx.y);            px.y = bf16_2(vx.z, vx.w);
    pm.x = bf16_2(vm.x * inv, vm.y * inv); pm.y = bf16_2(vm.z * inv, vm.w * inv);
    *(uint2*)&sAx[r * LDS_STRIDE + c] = px;   // ds_store_b64, 8B aligned
    *(uint2*)&sAg[r * LDS_STRIDE + c] = pm;
  }
  __syncthreads();

  const int wave  = t >> 5;
  const int lane  = t & 31;
  const int half  = lane >> 4;
  const int lr    = lane & 15;
  const int mtile = blockIdx.x * WAVES_PER_BLOCK + wave;
  if (mtile >= M_TILES) return;        // after the only barrier: safe

  // 16-bit A/B fragment per lane = two contiguous 8-elem K-runs:
  // k = kt*32 + half*8 + {0..7} and +16 -> two aligned ds_load_b128 each.
  v16bf Ag[4], Ax[4];
#pragma unroll
  for (int kt = 0; kt < 4; ++kt) {
    const int off = (wave * 16 + lr) * LDS_STRIDE + kt * 32 + half * 8;
    const v8bf* pg = (const v8bf*)(sAg + off);
    const v8bf* px = (const v8bf*)(sAx + off);
    ((v8bf*)&Ag[kt])[0] = pg[0];  ((v8bf*)&Ag[kt])[1] = pg[2];
    ((v8bf*)&Ax[kt])[0] = px[0];  ((v8bf*)&Ax[kt])[1] = px[2];
  }

  v8f acc[8];
#pragma unroll
  for (int nt = 0; nt < 8; ++nt) { v8f z = {}; acc[nt] = z; }

#pragma unroll
  for (int nt = 0; nt < 8; ++nt) {
#pragma unroll
    for (int kt = 0; kt < 4; ++kt) {
      const int boff = (nt * 16 + lr) * LDS_STRIDE + kt * 32 + half * 8;
      v16bf Bl, Br;
      const v8bf* pl = (const v8bf*)(sWl + boff);
      const v8bf* pr = (const v8bf*)(sWr + boff);
      ((v8bf*)&Bl)[0] = pl[0];  ((v8bf*)&Bl)[1] = pl[2];
      ((v8bf*)&Br)[0] = pr[0];  ((v8bf*)&Br)[1] = pr[2];
      acc[nt] = __builtin_amdgcn_wmma_f32_16x16x32_bf16(
          false, Ag[kt], false, Bl, (short)0, acc[nt], false, false);
      acc[nt] = __builtin_amdgcn_wmma_f32_16x16x32_bf16(
          false, Ax[kt], false, Br, (short)0, acc[nt], false, false);
    }
  }

  // Epilogue: C layout = VGPR i -> M = half*8 + i ; N = nt*16 + lane%16.
#pragma unroll
  for (int nt = 0; nt < 8; ++nt) {
    const float bv = bias[nt * 16 + lr];
#pragma unroll
    for (int i = 0; i < 8; ++i) {
      int M = mtile * 16 + half * 8 + i;
      out[M * (long long)D + nt * 16 + lr] = fmaxf(acc[nt][i] + bv, 0.0f);
    }
  }
}

// --------------------------------------------------------------------------
extern "C" void kernel_launch(void* const* d_in, const int* in_sizes, int n_in,
                              void* d_out, int out_size, void* d_ws, size_t ws_size,
                              hipStream_t stream) {
  const float*     x  = (const float*)d_in[0];
  const long long* ei = (const long long*)d_in[1];   // int64 [2][N_EDGES]
  const float*     Wl = (const float*)d_in[2];
  const float*     Wr = (const float*)d_in[3];
  const float*     b  = (const float*)d_in[4];
  float* out = (float*)d_out;

  float* msg = (float*)d_ws;                          // [N_NODES*D] fp32
  float* deg = msg + (size_t)N_NODES * D;             // [N_NODES]   fp32

  // 1) zero msg+deg (6,450,000 floats, divisible by 4)
  const int n4 = (N_NODES * D + N_NODES) / 4;
  k_zero_f4<<<(n4 + 255) / 256, 256, 0, stream>>>((float4*)d_ws, n4);

  // 2) scatter-add: one wave per edge
  k_scatter<<<(N_EDGES * 32) / 256, 256, 0, stream>>>(x, ei, msg, deg);

  // 3) fused normalize + dual BF16-WMMA GEMM + bias + ReLU
  const int grid = (M_TILES + WAVES_PER_BLOCK - 1) / WAVES_PER_BLOCK;  // 391
  k_sage_wmma<<<grid, 256, 0, stream>>>(x, msg, deg, Wl, Wr, b, out);
}
